// TWIN_33011118637610
// MI455X (gfx1250) — compile-verified, matrix operand order checked by reading, not run on hardware
//
#include <hip/hip_runtime.h>

// ---------------------------------------------------------------------------
// MI455X (gfx1250) implementation, round 3.
//   - NT GEMM: C[M,N] = A[M,128] * BT[N,128]^T, all bf16 via
//     v_wmma_f32_16x16x32_bf16 (8 WMMA per wave, full K staged in LDS).
//   - Staging uses gfx1250 async copy: global_load_async_to_lds_b128 +
//     s_wait_asynccnt (ISA 08_async_tensor §4).
//   - wave32: 256-thread blocks = 8 waves; each wave owns 16x32 of C.
//   - Bias loads hoisted out of the store loop (one load per column).
// ---------------------------------------------------------------------------

#define NTOK 4096   // B*S = 64*64
#define DDIM 128
#define KNBR 10
#define LDP  136    // padded LDS row length in bf16 (272B, 16B aligned)

typedef __bf16 v16bf __attribute__((ext_vector_type(16)));
typedef float  v8f   __attribute__((ext_vector_type(8)));

static __device__ __forceinline__ unsigned short f2bf(float f) {
    unsigned int u = __float_as_uint(f);
    u += 0x7FFFu + ((u >> 16) & 1u);          // round-to-nearest-even
    return (unsigned short)(u >> 16);
}

// gfx1250 async global->LDS copy, 16 bytes per enabled lane (ASYNCcnt).
static __device__ __forceinline__ void async_copy_b128(unsigned lds_off, const void* gaddr) {
    asm volatile("global_load_async_to_lds_b128 %0, %1, off"
                 :: "v"(lds_off), "v"(gaddr) : "memory");
}
static __device__ __forceinline__ void async_wait0() {
    asm volatile("s_wait_asynccnt 0x0" ::: "memory");
}

// ---- embedding gather + sum over code dim; emit f32 and bf16
__global__ void embed_sum_kernel(const int* __restrict__ seq,
                                 const float* __restrict__ table,
                                 float* __restrict__ h32,
                                 unsigned short* __restrict__ hbf,
                                 int C) {
    int row = blockIdx.x;
    int d   = threadIdx.x;
    const int* s = seq + row * C;
    float acc = 0.f;
    for (int c = 0; c < C; ++c)
        acc += table[(size_t)s[c] * DDIM + d];
    h32[(size_t)row * DDIM + d] = acc;
    hbf[(size_t)row * DDIM + d] = f2bf(acc);
}

// ---- NT bf16 WMMA GEMM: C[M,N] = A[M,128] * BT[N,128]^T (+bias)(+relu)
// Block: 256 threads / 8 waves, 64x64 C tile, full K=128 staged in LDS.
__global__ void gemm_bf16_nt_kernel(const unsigned short* __restrict__ A,
                                    const unsigned short* __restrict__ BT,
                                    const float* __restrict__ bias,
                                    float* __restrict__ C,
                                    unsigned short* __restrict__ Cbf,
                                    int M, int N, int relu) {
    __shared__ unsigned short As[64 * LDP];
    __shared__ unsigned short Bs[64 * LDP];

    int tid  = threadIdx.x;       // 0..255
    int lane = tid & 31;
    int wave = tid >> 5;          // 0..7
    int half = lane >> 4;         // 0/1
    int sub  = lane & 15;
    int mBase = blockIdx.y * 64;
    int nBase = blockIdx.x * 64;

    // ---- stage A (always in range: M % 64 == 0): async global->LDS
    #pragma unroll
    for (int pass = 0; pass < 4; ++pass) {
        int e   = pass * 256 + tid;         // chunk id, 16B chunks
        int row = e >> 4, c = e & 15;       // 16 chunks per 128-elem row
        unsigned dst = (unsigned)(size_t)&As[row * LDP + c * 8];
        async_copy_b128(dst, A + (size_t)(mBase + row) * DDIM + c * 8);
    }
    // ---- stage BT rows (zero-fill rows >= N)
    #pragma unroll
    for (int pass = 0; pass < 4; ++pass) {
        int e   = pass * 256 + tid;
        int row = e >> 4, c = e & 15;
        if (nBase + row < N) {
            unsigned dst = (unsigned)(size_t)&Bs[row * LDP + c * 8];
            async_copy_b128(dst, BT + (size_t)(nBase + row) * DDIM + c * 8);
        } else {
            uint4 z = make_uint4(0u, 0u, 0u, 0u);
            *(uint4*)(&Bs[row * LDP + c * 8]) = z;
        }
    }
    async_wait0();
    __syncthreads();

    // ---- compute: each wave: A tile rows [wm,wm+16), B tiles [wn,wn+16),[wn+16,wn+32)
    int wm = (wave >> 1) * 16;
    int wn = (wave & 1) * 32;
    const unsigned short* ar  = &As[(wm + sub) * LDP];
    const unsigned short* br0 = &Bs[(wn + sub) * LDP];
    const unsigned short* br1 = &Bs[(wn + 16 + sub) * LDP];

    v8f acc0 = {}, acc1 = {};
    #pragma unroll
    for (int s = 0; s < 4; ++s) {         // K steps of 32
        union { v16bf v; unsigned int u[8]; } fa, fb0, fb1;
        #pragma unroll
        for (int p = 0; p < 8; ++p) {     // ISA 7.12.2 16-bit A/B pair layout
            int kk = s * 32 + (p < 4 ? 2 * p : 16 + 2 * (p - 4)) + 8 * half;
            fa.u[p]  = *(const unsigned int*)(ar + kk);
            fb0.u[p] = *(const unsigned int*)(br0 + kk);
            fb1.u[p] = *(const unsigned int*)(br1 + kk);
        }
        acc0 = __builtin_amdgcn_wmma_f32_16x16x32_bf16(
                   false, fa.v, false, fb0.v, (short)0, acc0, false, false);
        acc1 = __builtin_amdgcn_wmma_f32_16x16x32_bf16(
                   false, fa.v, false, fb1.v, (short)0, acc1, false, false);
    }

    // ---- epilogue: hoist per-column state (bias, bounds) out of the r loop
    int col0 = nBase + wn + sub;
    int col1 = col0 + 16;
    bool in0 = col0 < N, in1 = col1 < N;
    float b0 = 0.f, b1 = 0.f;
    if (bias) {
        if (in0) b0 = bias[col0];
        if (in1) b1 = bias[col1];
    }

    // C/D layout: VGPR r -> row = r + 8*half; col = lane%16
    #pragma unroll
    for (int r = 0; r < 8; ++r) {
        int row = mBase + wm + r + 8 * half;
        float v0 = acc0[r] + b0;
        float v1 = acc1[r] + b1;
        if (relu) { v0 = fmaxf(v0, 0.f); v1 = fmaxf(v1, 0.f); }
        if (in0) {
            C[(size_t)row * N + col0] = v0;
            if (Cbf) Cbf[(size_t)row * N + col0] = f2bf(v0);
        }
        if (in1) {
            C[(size_t)row * N + col1] = v1;
            if (Cbf) Cbf[(size_t)row * N + col1] = f2bf(v1);
        }
    }
}

// ---- per-row top-K over the sim matrix (self masked)
__global__ void topk_kernel(const float* __restrict__ sim,
                            float* __restrict__ vals,
                            int* __restrict__ inds) {
    __shared__ float sv[256 * KNBR];
    __shared__ int   si[256 * KNBR];
    int row = blockIdx.x;
    int tid = threadIdx.x;
    float lv[KNBR]; int li[KNBR];
    #pragma unroll
    for (int k = 0; k < KNBR; ++k) { lv[k] = -__builtin_inff(); li[k] = 0; }
    const float* srow = sim + (size_t)row * NTOK;
    for (int c = tid; c < NTOK; c += 256) {
        if (c == row) continue;               // mask self
        float v = srow[c];
        if (v > lv[KNBR - 1]) {
            int pos = KNBR - 1;
            while (pos > 0 && lv[pos - 1] < v) {
                lv[pos] = lv[pos - 1]; li[pos] = li[pos - 1]; --pos;
            }
            lv[pos] = v; li[pos] = c;
        }
    }
    for (int k = 0; k < KNBR; ++k) {
        sv[tid * KNBR + k] = lv[k];
        si[tid * KNBR + k] = li[k];
    }
    __syncthreads();
    if (tid == 0) {
        for (int k = 0; k < KNBR; ++k) {
            float best = -__builtin_inff(); int bi = 0;
            for (int e = 0; e < 256 * KNBR; ++e)
                if (sv[e] > best) { best = sv[e]; bi = e; }
            vals[(size_t)row * KNBR + k] = best;
            inds[(size_t)row * KNBR + k] = si[bi];
            sv[bi] = -__builtin_inff();
        }
    }
}

// ---- softmax over K neighbor scores, weighted gather + residual
__global__ void attend_kernel(const float* __restrict__ h32,
                              const float* __restrict__ vals,
                              const int* __restrict__ inds,
                              float* __restrict__ hb32,
                              unsigned short* __restrict__ hbbf) {
    __shared__ float w[KNBR];
    __shared__ int   id[KNBR];
    int row = blockIdx.x, d = threadIdx.x;
    if (d == 0) {
        float m = -__builtin_inff();
        for (int k = 0; k < KNBR; ++k) m = fmaxf(m, vals[(size_t)row * KNBR + k]);
        float s = 0.f;
        for (int k = 0; k < KNBR; ++k) {
            float e = __expf(vals[(size_t)row * KNBR + k] - m);
            w[k] = e; s += e;
        }
        for (int k = 0; k < KNBR; ++k) w[k] /= s;
    }
    if (d < KNBR) id[d] = inds[(size_t)row * KNBR + d];
    __syncthreads();
    float acc = h32[(size_t)row * DDIM + d];
    for (int k = 0; k < KNBR; ++k)
        acc += w[k] * h32[(size_t)id[k] * DDIM + d];
    hb32[(size_t)row * DDIM + d] = acc;
    hbbf[(size_t)row * DDIM + d] = f2bf(acc);
}

// ---- reparameterize: z = (mu[+mu_diag]) + exp(0.5*(lv[+lv_diag]))*eps
__global__ void z_kernel(const float* __restrict__ mu_n,
                         const float* __restrict__ lv_n,
                         const float* __restrict__ mu_d,
                         const float* __restrict__ lv_d,
                         const float* __restrict__ eps,
                         unsigned short* __restrict__ zbf, int addDiag) {
    int i = blockIdx.x * blockDim.x + threadIdx.x;
    float m = mu_n[i], l = lv_n[i];
    if (addDiag) { m += mu_d[i]; l += lv_d[i]; }
    float z = m + __expf(0.5f * l) * eps[i];
    zbf[i] = f2bf(z);
}

// ---- transpose-convert weights: W[128,N] f32 -> WT[N,128] bf16
__global__ void wtrans_bf16_kernel(const float* __restrict__ W,
                                   unsigned short* __restrict__ WT, int N) {
    __shared__ float t[16][17];
    int nb = blockIdx.x * 16, kb = blockIdx.y * 16;
    int tx = threadIdx.x, ty = threadIdx.y;   // 16x16
    int n = nb + tx;
    t[ty][tx] = (n < N) ? W[(size_t)(kb + ty) * N + n] : 0.f;
    __syncthreads();
    int n2 = nb + ty;
    if (n2 < N) WT[(size_t)n2 * DDIM + kb + tx] = f2bf(t[tx][ty]);
}

__global__ void concat_copy_kernel(const float* __restrict__ src,
                                   float* __restrict__ dst, int colOff) {
    int row = blockIdx.x, d = threadIdx.x;
    dst[(size_t)row * (4 * DDIM) + colOff + d] = src[(size_t)row * DDIM + d];
}

// ---------------------------------------------------------------------------
extern "C" void kernel_launch(void* const* d_in, const int* in_sizes, int n_in,
                              void* d_out, int out_size, void* d_ws, size_t ws_size,
                              hipStream_t stream) {
    (void)in_sizes; (void)n_in; (void)out_size; (void)ws_size;
    static const int VOC[4] = {10000, 2000, 600, 1500};

    // Input order: 4 seqs, params in insertion order, then eps (diag..proc).
    const int*   seq[4]; const float* emb[4];
    for (int i = 0; i < 4; ++i) { seq[i] = (const int*)d_in[i]; emb[i] = (const float*)d_in[4 + i]; }
    const float* We_mu = (const float*)d_in[8];
    const float* be_mu = (const float*)d_in[9];
    const float* We_lv = (const float*)d_in[10];
    const float* be_lv = (const float*)d_in[11];
    const float* Wd    = (const float*)d_in[12];
    const float* bd    = (const float*)d_in[13];
    const float *W1[4], *b1[4], *W2[4], *b2[4];
    for (int i = 0; i < 4; ++i) {
        W1[i] = (const float*)d_in[14 + 4 * i];
        b1[i] = (const float*)d_in[15 + 4 * i];
        W2[i] = (const float*)d_in[16 + 4 * i];
        b2[i] = (const float*)d_in[17 + 4 * i];
    }
    const float* eps[4];
    for (int i = 0; i < 4; ++i) eps[i] = (const float*)d_in[30 + i];

    // Workspace layout (deterministic bump allocator)
    char* base = (char*)d_ws;
    size_t off = 0;
    auto alloc = [&](size_t bytes) -> char* {
        char* p = base + off;
        off = (off + bytes + 255) & ~(size_t)255;
        return p;
    };
    float*          sim  = (float*)alloc((size_t)NTOK * NTOK * 4);
    float*          h32  = (float*)alloc((size_t)NTOK * DDIM * 4);
    float*          hb32 = (float*)alloc((size_t)NTOK * DDIM * 4);
    unsigned short* hbf  = (unsigned short*)alloc((size_t)NTOK * DDIM * 2);
    unsigned short* hbbf = (unsigned short*)alloc((size_t)NTOK * DDIM * 2);
    float*          vals = (float*)alloc((size_t)NTOK * KNBR * 4);
    int*            inds = (int*)alloc((size_t)NTOK * KNBR * 4);
    float *mu[4], *lv[4];
    for (int i = 0; i < 4; ++i) mu[i] = (float*)alloc((size_t)NTOK * DDIM * 4);
    for (int i = 0; i < 4; ++i) lv[i] = (float*)alloc((size_t)NTOK * DDIM * 4);
    unsigned short* zbf = (unsigned short*)alloc((size_t)NTOK * DDIM * 2);
    float*          v32 = (float*)alloc((size_t)NTOK * DDIM * 4);
    unsigned short* vbf = (unsigned short*)alloc((size_t)NTOK * DDIM * 2);
    float*          m32 = (float*)alloc((size_t)NTOK * DDIM * 4);
    unsigned short* mbf = (unsigned short*)alloc((size_t)NTOK * DDIM * 2);
    unsigned short* WmuT = (unsigned short*)alloc(DDIM * DDIM * 2);
    unsigned short* WlvT = (unsigned short*)alloc(DDIM * DDIM * 2);
    unsigned short* WdT  = (unsigned short*)alloc(DDIM * DDIM * 2);
    unsigned short *W1T[4], *W2T[4];
    for (int i = 0; i < 4; ++i) W1T[i] = (unsigned short*)alloc(DDIM * DDIM * 2);
    for (int i = 0; i < 4; ++i) W2T[i] = (unsigned short*)alloc((size_t)DDIM * VOC[i] * 2);

    auto wtrans = [&](const float* s, unsigned short* d, int N) {
        wtrans_bf16_kernel<<<dim3((N + 15) / 16, DDIM / 16), dim3(16, 16), 0, stream>>>(s, d, N);
    };
    wtrans(We_mu, WmuT, DDIM);
    wtrans(We_lv, WlvT, DDIM);
    wtrans(Wd,    WdT,  DDIM);
    for (int i = 0; i < 4; ++i) wtrans(W1[i], W1T[i], DDIM);
    for (int i = 0; i < 4; ++i) wtrans(W2[i], W2T[i], VOC[i]);

    auto gemm = [&](const unsigned short* A, const unsigned short* BT,
                    const float* bias, float* C, unsigned short* Cb,
                    int M, int N, int relu) {
        dim3 grid((N + 63) / 64, M / 64);
        gemm_bf16_nt_kernel<<<grid, 256, 0, stream>>>(A, BT, bias, C, Cb, M, N, relu);
    };

    // Phase 1: per modality -> h, sim (A=h, BT=h), topK, attend, mu/lv
    for (int i = 0; i < 4; ++i) {
        embed_sum_kernel<<<NTOK, DDIM, 0, stream>>>(seq[i], emb[i], h32, hbf, 20);
        gemm(hbf, hbf, nullptr, sim, nullptr, NTOK, NTOK, 0);
        topk_kernel<<<NTOK, 256, 0, stream>>>(sim, vals, inds);
        attend_kernel<<<NTOK, DDIM, 0, stream>>>(h32, vals, inds, hb32, hbbf);
        gemm(hbbf, WmuT, be_mu, mu[i], nullptr, NTOK, DDIM, 0);
        gemm(hbbf, WlvT, be_lv, lv[i], nullptr, NTOK, DDIM, 0);
    }

    // Output offsets: logits[4], concat_mu, concat_lv
    size_t logitOff[4]; size_t run = 0;
    for (int i = 0; i < 4; ++i) { logitOff[i] = run; run += (size_t)NTOK * VOC[i]; }
    float* out = (float*)d_out;
    size_t muOff = run;
    size_t lvOff = run + (size_t)NTOK * 4 * DDIM;

    // Phase 2: reparameterize -> decode -> MLP -> logits
    for (int i = 0; i < 4; ++i) {
        z_kernel<<<NTOK, DDIM, 0, stream>>>(mu[i], lv[i], mu[0], lv[0], eps[i], zbf, i != 0);
        gemm(zbf, WdT, bd, v32, vbf, NTOK, DDIM, 0);
        gemm(vbf, W1T[i], b1[i], m32, mbf, NTOK, DDIM, 1);
        gemm(mbf, W2T[i], b2[i], out + logitOff[i], nullptr, NTOK, VOC[i], 0);
    }

    // Concatenate mu / lv
    for (int i = 0; i < 4; ++i) {
        concat_copy_kernel<<<NTOK, DDIM, 0, stream>>>(mu[i], out + muOff, i * DDIM);
        concat_copy_kernel<<<NTOK, DDIM, 0, stream>>>(lv[i], out + lvOff, i * DDIM);
    }
}
